// Model_4372276707541
// MI455X (gfx1250) — matrix operand order repaired and checked by reading, not compiled, not run on hardware
//
#include <hip/hip_runtime.h>

typedef __attribute__((ext_vector_type(16))) _Float16 v16h;
typedef __attribute__((ext_vector_type(8)))  _Float16 v8h;
typedef __attribute__((ext_vector_type(8)))  float    v8f;

#define WD    50
#define PD    5
#define HID   230
#define REL   53
#define LQ    120
#define BAGS  512
#define NSENT 4096
#define QL    8
#define FEAT  160        // 3*WD + 2*PD
#define KTOT  480        // 3*FEAT
#define HPAD  256        // HID padded to 16 WMMA n-tiles (branch-free wave split)
#define XROWS 130        // rows -1..128 of the conv input (zero padded)
#define XSTR  168        // LDS row stride in halves (16B aligned, bank-conflict free)

// ---------------------------------------------------------------------------
// Kernel 1: build per-bag query embeddings q[b][j][d] (slots QL-4 / QL-2 get
// entity-mention means) plus qw[b][j] = q . w_q + b_q.
// ---------------------------------------------------------------------------
__global__ __launch_bounds__(64) void qbuild_kernel(
    const float* __restrict__ Wword, const int* __restrict__ E1,
    const int* __restrict__ E2, const int* __restrict__ Q,
    const float* __restrict__ w_q, const float* __restrict__ b_q,
    float* __restrict__ qbuf, float* __restrict__ qw)
{
  __shared__ float qsh[QL * WD];
  const int b = blockIdx.x, tid = threadIdx.x;
  for (int i = tid; i < QL * WD; i += 64) {
    int j = i / WD, d = i % WD;
    float v;
    if (j == QL - 4)
      v = 0.5f * (Wword[(size_t)E1[b*2]*WD + d] + Wword[(size_t)E1[b*2+1]*WD + d]);
    else if (j == QL - 2)
      v = 0.5f * (Wword[(size_t)E2[b*2]*WD + d] + Wword[(size_t)E2[b*2+1]*WD + d]);
    else
      v = Wword[(size_t)Q[b*QL + j]*WD + d];
    qsh[i] = v;
    qbuf[(size_t)b*QL*WD + i] = v;
  }
  __syncthreads();
  if (tid < QL) {
    float a = b_q[0];
    for (int d = 0; d < WD; ++d) a += qsh[tid*WD + d] * w_q[d];
    qw[b*QL + tid] = a;
  }
}

// ---------------------------------------------------------------------------
// Kernel 2: conv_W [3][160][230] fp32  ->  WhT fp16 column-major [256][480]
// (columns 230..255 zero padded) so B-fragments are contiguous 16-half runs
// per lane and the 16 n-tiles split 2-per-wave with no divergence.
// ---------------------------------------------------------------------------
__global__ __launch_bounds__(256) void wconv_kernel(
    const float* __restrict__ convW, _Float16* __restrict__ WhT)
{
  int i = blockIdx.x * 256 + threadIdx.x;
  if (i >= HPAD * KTOT) return;
  int c2 = i / KTOT, K = i % KTOT;
  int kt = K / FEAT, c = K % FEAT;
  float v = (c2 < HID) ? convW[(size_t)(kt*FEAT + c)*HID + c2] : 0.0f;
  WhT[i] = (_Float16)v;
}

// ---------------------------------------------------------------------------
// Kernel 3 (fused, 1 block = 1 sentence, 256 threads = 8 waves):
//   phase A: gather embeddings, BiDAF attention, build fp16 feature rows in LDS
//   phase B: conv1d(k=3) as tiled WMMA f32_16x16x32_f16 GEMM, relu(max over l)
// ---------------------------------------------------------------------------
__global__ __launch_bounds__(256) void sent_kernel(
    const int* __restrict__ X, const int* __restrict__ Xp1, const int* __restrict__ Xp2,
    const float* __restrict__ Wword, const float* __restrict__ Wp1, const float* __restrict__ Wp2,
    const float* __restrict__ w_s, const float* __restrict__ b_s,
    const float* __restrict__ w_sq, const float* __restrict__ b_sq,
    const float* __restrict__ qbuf, const float* __restrict__ qw,
    const _Float16* __restrict__ WhT, const float* __restrict__ conv_b,
    float* __restrict__ enc)
{
  __shared__ _Float16 xh[XROWS * XSTR];   // conv input rows -1..128, fp16
  __shared__ float qloc[QL * WD];
  __shared__ float qs[QL * WD];           // q * w_sq
  __shared__ float hbuf[LQ * QL];         // h then softmax(a)
  __shared__ float sw[LQ];
  __shared__ float hmax[LQ];              // row max then b_att
  __shared__ float q2s[WD];
  __shared__ float qwv[QL];
  __shared__ float scal[2];

  const int tid = threadIdx.x;
  const int n   = blockIdx.x;
  const int bag = n >> 3;

  // Warm L2 with the fp16 conv weights (246 KB, reused by all 4096 blocks).
  for (int off = tid * 128; off < HPAD * KTOT * 2; off += 256 * 128)
    __builtin_prefetch((const char*)WhT + off, 0, 1);

  // Zero the pad rows of xh (row 0 == l=-1, rows 121..129 == l=120..128).
  for (int i = tid; i < 10 * XSTR; i += 256) {
    int r = i / XSTR, c = i % XSTR;
    int row = (r == 0) ? 0 : (120 + r);
    xh[row * XSTR + c] = (_Float16)0.0f;
  }

  // Gather word / position embeddings (s -> cols 0..49, p1/p2 -> 150..159).
  for (int i = tid; i < LQ * WD; i += 256) {
    int l = i / WD, d = i % WD;
    xh[(l+1)*XSTR + d] = (_Float16)Wword[(size_t)X[n*LQ + l]*WD + d];
  }
  for (int i = tid; i < LQ * PD; i += 256) {
    int l = i / PD, d = i % PD;
    xh[(l+1)*XSTR + 150 + d] = (_Float16)Wp1[(size_t)Xp1[n*LQ + l]*PD + d];
    xh[(l+1)*XSTR + 155 + d] = (_Float16)Wp2[(size_t)Xp2[n*LQ + l]*PD + d];
  }
  for (int i = tid; i < QL * WD; i += 256) {
    float v = qbuf[(size_t)bag*QL*WD + i];
    qloc[i] = v;
    qs[i]   = v * w_sq[i % WD];
  }
  if (tid < QL) qwv[tid] = qw[bag*QL + tid];
  __syncthreads();

  // sw[l] = s[l] . w_s + b_s
  if (tid < LQ) {
    float a = b_s[0];
    for (int d = 0; d < WD; ++d) a += (float)xh[(tid+1)*XSTR + d] * w_s[d];
    sw[tid] = a;
  }
  __syncthreads();

  // h[l][j] = sw[l] + qw[j] + b_sq + sum_d s*w_sq*q
  const float bsq = b_sq[0];
  for (int i = tid; i < LQ * QL; i += 256) {
    int l = i >> 3, j = i & 7;
    float a = 0.0f;
    for (int d = 0; d < WD; ++d) a += (float)xh[(l+1)*XSTR + d] * qs[j*WD + d];
    hbuf[i] = a + sw[l] + qwv[j] + bsq;
  }
  __syncthreads();

  // softmax over j (8) in place; keep raw row max for b_att.
  if (tid < LQ) {
    float mx = -1e30f;
    for (int j = 0; j < QL; ++j) mx = fmaxf(mx, hbuf[tid*QL + j]);
    float s = 0.0f;
    for (int j = 0; j < QL; ++j) { float e = __expf(hbuf[tid*QL + j] - mx); hbuf[tid*QL + j] = e; s += e; }
    float inv = 1.0f / s;
    for (int j = 0; j < QL; ++j) hbuf[tid*QL + j] *= inv;
    hmax[tid] = mx;
  }
  __syncthreads();

  // b_att = softmax over l of hmax (wave-0 shfl reduction).
  if (tid < 32) {
    float lm = -1e30f;
    for (int l = tid; l < LQ; l += 32) lm = fmaxf(lm, hmax[l]);
    for (int o = 16; o > 0; o >>= 1) lm = fmaxf(lm, __shfl_xor(lm, o, 32));
    float ls = 0.0f;
    for (int l = tid; l < LQ; l += 32) ls += __expf(hmax[l] - lm);
    for (int o = 16; o > 0; o >>= 1) ls += __shfl_xor(ls, o, 32);
    if (tid == 0) { scal[0] = lm; scal[1] = 1.0f / ls; }
  }
  __syncthreads();
  if (tid < LQ) hmax[tid] = __expf(hmax[tid] - scal[0]) * scal[1];
  __syncthreads();

  // q2s[d] = sum_l b_att[l] * s[l][d]
  if (tid < WD) {
    float a = 0.0f;
    for (int l = 0; l < LQ; ++l) a += hmax[l] * (float)xh[(l+1)*XSTR + tid];
    q2s[tid] = a;
  }
  __syncthreads();

  // features: cols 50..99 = s*s2q, cols 100..149 = s*q2s
  for (int i = tid; i < LQ * WD; i += 256) {
    int l = i / WD, d = i % WD;
    float sv = (float)xh[(l+1)*XSTR + d];
    float s2 = 0.0f;
    for (int j = 0; j < QL; ++j) s2 += hbuf[l*QL + j] * qloc[j*WD + d];
    xh[(l+1)*XSTR + 50  + d] = (_Float16)(sv * s2);
    xh[(l+1)*XSTR + 100 + d] = (_Float16)(sv * q2s[d]);
  }
  __syncthreads();

  // ---------------- WMMA conv: y[l,h] = sum_K A[l,K] * W[K,h] -------------
  // A[l, kk*32+j] = xh[l + kk/5][(kk%5)*32 + j]   (160 = 5*32, tap-aligned)
  // 16 n-tiles over 8 waves: every wave owns {wave, wave+8} -> no divergence.
  const int wave = tid >> 5, lane = tid & 31;
  const int m = lane & 15, half = lane >> 4;
  const int nt0 = wave, nt1 = wave + 8;
  const int col0 = nt0 * 16 + m;
  const int col1 = nt1 * 16 + m;

  float run0 = -1e30f, run1 = -1e30f;

  for (int mtb = 0; mtb < 2; ++mtb) {         // 8 m-tiles in 2 blocks of 4
    v8f acc0[4], acc1[4];
    #pragma unroll
    for (int mi = 0; mi < 4; ++mi)
      #pragma unroll
      for (int e = 0; e < 8; ++e) { acc0[mi][e] = 0.0f; acc1[mi][e] = 0.0f; }

    for (int kk = 0; kk < 15; ++kk) {
      const int kt = kk / 5, cb = (kk % 5) * 32;
      // B fragments: lane = 16*half + col, element e <-> K = kk*32 + half*16 + e
      const _Float16* bp0 = WhT + (size_t)col0 * KTOT + kk*32 + half*16;
      const _Float16* bp1 = WhT + (size_t)col1 * KTOT + kk*32 + half*16;
      v8h b0lo = *(const v8h*)bp0, b0hi = *(const v8h*)(bp0 + 8);
      v8h b1lo = *(const v8h*)bp1, b1hi = *(const v8h*)(bp1 + 8);
      v16h b0, b1;
      #pragma unroll
      for (int e = 0; e < 8; ++e) {
        b0[e] = b0lo[e]; b0[8+e] = b0hi[e];
        b1[e] = b1lo[e]; b1[8+e] = b1hi[e];
      }
      #pragma unroll
      for (int mi = 0; mi < 4; ++mi) {
        // A fragment: lane m holds row m; elements 0..15 <-> K half*16+e
        // (contiguous, matching B).
        const int row = (mtb*4 + mi) * 16 + m + kt;
        const _Float16* ap = &xh[row * XSTR + cb + half*16];
        v8h alo = *(const v8h*)ap, ahi = *(const v8h*)(ap + 8);
        v16h a;
        #pragma unroll
        for (int e = 0; e < 8; ++e) { a[e] = alo[e]; a[8+e] = ahi[e]; }
        acc0[mi] = __builtin_amdgcn_wmma_f32_16x16x32_f16(
            false, a, false, b0, (short)0, acc0[mi], false, false);
        acc1[mi] = __builtin_amdgcn_wmma_f32_16x16x32_f16(
            false, a, false, b1, (short)0, acc1[mi], false, false);
      }
    }
    #pragma unroll
    for (int mi = 0; mi < 4; ++mi)
      #pragma unroll
      for (int e = 0; e < 8; ++e) {
        run0 = fmaxf(run0, acc0[mi][e]);
        run1 = fmaxf(run1, acc1[mi][e]);
      }
  }
  // lane L and L^16 hold rows 0-7 vs 8-15 of each tile for the same column
  run0 = fmaxf(run0, __shfl_xor(run0, 16, 32));
  run1 = fmaxf(run1, __shfl_xor(run1, 16, 32));

  if (half == 0) {
    int g0 = nt0 * 16 + m;
    int g1 = nt1 * 16 + m;
    if (g0 < HID) enc[(size_t)n*HID + g0] = fmaxf(run0 + conv_b[g0], 0.0f);
    if (g1 < HID) enc[(size_t)n*HID + g1] = fmaxf(run1 + conv_b[g1], 0.0f);
  }
}

// ---------------------------------------------------------------------------
// Kernel 4: bag self-attention + relation scoring (1 block = 1 bag).
// ---------------------------------------------------------------------------
__global__ __launch_bounds__(64) void bag_kernel(
    const float* __restrict__ enc, const float* __restrict__ relW,
    const float* __restrict__ relB, const int* __restrict__ XRel,
    float* __restrict__ out)
{
  __shared__ float bagv[8 * 232];
  __shared__ float bag2[8 * 232];
  __shared__ float att[64];
  __shared__ float sc[8];
  __shared__ float rep[232];
  const int b = blockIdx.x, tid = threadIdx.x;

  for (int i = tid; i < 8 * HID; i += 64) {
    int r = i / HID, d = i % HID;
    bagv[r*232 + d] = enc[(size_t)(b*8 + r)*HID + d];
  }
  __syncthreads();
  {
    int r = tid >> 3, c = tid & 7;
    float a = 0.0f;
    for (int d = 0; d < HID; ++d) a += bagv[r*232 + d] * bagv[c*232 + d];
    att[tid] = a * 0.0659380473f;   // 1/sqrt(230)
  }
  __syncthreads();
  if (tid < 8) {
    float mx = -1e30f;
    for (int j = 0; j < 8; ++j) mx = fmaxf(mx, att[tid*8 + j]);
    float s = 0.0f;
    for (int j = 0; j < 8; ++j) { float e = __expf(att[tid*8+j] - mx); att[tid*8+j] = e; s += e; }
    float inv = 1.0f / s;
    for (int j = 0; j < 8; ++j) att[tid*8+j] *= inv;
  }
  __syncthreads();
  for (int i = tid; i < 8 * HID; i += 64) {
    int r = i / HID, d = i % HID;
    float a = 0.0f;
    for (int j = 0; j < 8; ++j) a += att[r*8 + j] * bagv[j*232 + d];
    bag2[r*232 + d] = a;
  }
  __syncthreads();
  if (tid < 8) {
    const float* rv = relW + (size_t)XRel[b]*HID;
    float a = 0.0f;
    for (int d = 0; d < HID; ++d) a += bag2[tid*232 + d] * rv[d];
    sc[tid] = a;
  }
  __syncthreads();
  if (tid == 0) {
    float mx = -1e30f;
    for (int j = 0; j < 8; ++j) mx = fmaxf(mx, sc[j]);
    float s = 0.0f;
    for (int j = 0; j < 8; ++j) { float e = __expf(sc[j] - mx); sc[j] = e; s += e; }
    float inv = 1.0f / s;
    for (int j = 0; j < 8; ++j) sc[j] *= inv;
  }
  __syncthreads();
  for (int d = tid; d < HID; d += 64) {
    float a = 0.0f;
    for (int j = 0; j < 8; ++j) a += sc[j] * bag2[j*232 + d];
    rep[d] = a;
  }
  __syncthreads();
  for (int r = tid; r < REL; r += 64) {
    const float* wr = relW + (size_t)r*HID;
    float a = relB[r];
    for (int d = 0; d < HID; ++d) a += rep[d] * wr[d];
    out[(size_t)b*REL + r] = a;
  }
}

// ---------------------------------------------------------------------------
extern "C" void kernel_launch(void* const* d_in, const int* in_sizes, int n_in,
                              void* d_out, int out_size, void* d_ws, size_t ws_size,
                              hipStream_t stream) {
  const int*   X     = (const int*)d_in[0];
  const int*   Xp1   = (const int*)d_in[1];
  const int*   Xp2   = (const int*)d_in[2];
  const int*   E1    = (const int*)d_in[3];
  const int*   E2    = (const int*)d_in[4];
  // d_in[5..9]: X_Mask / X_Len / X_Scope / Scope_ent1 / Scope_ent2 (unused)
  const int*   Q     = (const int*)d_in[10];
  const int*   XRel  = (const int*)d_in[11];
  const float* Wword = (const float*)d_in[12];
  const float* Wp1   = (const float*)d_in[13];
  const float* Wp2   = (const float*)d_in[14];
  const float* w_s   = (const float*)d_in[15];
  const float* b_s   = (const float*)d_in[16];
  const float* w_q   = (const float*)d_in[17];
  const float* b_q   = (const float*)d_in[18];
  const float* w_sq  = (const float*)d_in[19];
  const float* b_sq  = (const float*)d_in[20];
  const float* convW = (const float*)d_in[21];
  const float* convB = (const float*)d_in[22];
  const float* relW  = (const float*)d_in[23];
  const float* relB  = (const float*)d_in[24];
  float* out = (float*)d_out;

  // Workspace layout (256-B aligned offsets), ~4.85 MB total:
  char* ws = (char*)d_ws;
  float*    qbuf = (float*)(ws + 0);          // 512*8*50*4  = 819200
  float*    qw   = (float*)(ws + 819200);     // 512*8*4     =  16384
  _Float16* WhT  = (_Float16*)(ws + 835584);  // 256*480*2   = 245760
  float*    enc  = (float*)(ws + 1081344);    // 4096*230*4  = 3768320

  qbuild_kernel<<<BAGS, 64, 0, stream>>>(Wword, E1, E2, Q, w_q, b_q, qbuf, qw);
  wconv_kernel<<<(HPAD*KTOT + 255)/256, 256, 0, stream>>>(convW, WhT);
  sent_kernel<<<NSENT, 256, 0, stream>>>(X, Xp1, Xp2, Wword, Wp1, Wp2,
                                         w_s, b_s, w_sq, b_sq,
                                         qbuf, qw, WhT, convB, enc);
  bag_kernel<<<BAGS, 64, 0, stream>>>(enc, relW, relB, XRel, out);
}
